// Attn_28741921145207
// MI455X (gfx1250) — matrix operand order repaired and checked by reading, not compiled, not run on hardware
//
#include <hip/hip_runtime.h>

typedef __attribute__((ext_vector_type(16))) _Float16 v16h;
typedef __attribute__((ext_vector_type(8)))  _Float16 v8h;
typedef __attribute__((ext_vector_type(8)))  float    v8f;

#define BB 2
#define NN 2048
#define CC 1024
#define HH 16
#define DD 64
#define SCALE 0.125f

static __device__ inline v8h ld8f2h(const float* p) {
  float4 a = ((const float4*)p)[0];
  float4 b = ((const float4*)p)[1];
  v8h r;
  r[0] = (_Float16)a.x; r[1] = (_Float16)a.y; r[2] = (_Float16)a.z; r[3] = (_Float16)a.w;
  r[4] = (_Float16)b.x; r[5] = (_Float16)b.y; r[6] = (_Float16)b.z; r[7] = (_Float16)b.w;
  return r;
}

static __device__ inline v8h ld8h(const _Float16* p) { return *(const v8h*)p; }

static __device__ inline v16h cat16(v8h lo, v8h hi) {
  v16h r;
#pragma unroll
  for (int i = 0; i < 8; i++) { r[i] = lo[i]; r[8 + i] = hi[i]; }
  return r;
}

static __device__ inline v16h ld16h(const _Float16* p) {
  return cat16(ld8h(p), ld8h(p + 8));
}

static __device__ inline v8f zero8() {
  v8f z;
#pragma unroll
  for (int i = 0; i < 8; i++) z[i] = 0.f;
  return z;
}

static __device__ inline v8f wmma16(v16h a, v16h b, v8f c) {
  return __builtin_amdgcn_wmma_f32_16x16x32_f16(
      /*neg_a=*/false, a, /*neg_b=*/false, b,
      /*c_mod=*/(short)0, c, /*reuse_a=*/false, /*reuse_b=*/false);
}

// ---------------------------------------------------------------------------
// Kernel 0: one-time fp32 -> f16 conversion (8 elements per thread).
// ---------------------------------------------------------------------------
__global__ __launch_bounds__(256) void f32_to_f16_kernel(
    const float* __restrict__ src, _Float16* __restrict__ dst, int n8)
{
  const int t = blockIdx.x * 256 + threadIdx.x;
  if (t < n8) {
    *(v8h*)(dst + (size_t)t * 8) = ld8f2h(src + (size_t)t * 8);
  }
}

// ---------------------------------------------------------------------------
// Kernel 1: QKV projection. xh[BN,C](f16) @ wh[3C,C]^T(f16) + b
//           -> q/k/v f16 [B,H,N,D].  Each wave: 2x2 block of 16x16 tiles
//           (4 WMMAs per k-step from 4 fragment loads).
// ---------------------------------------------------------------------------
__global__ __launch_bounds__(256) void qkv_gemm_kernel(
    const _Float16* __restrict__ xh, const _Float16* __restrict__ wh,
    const float* __restrict__ bias,
    _Float16* __restrict__ qh, _Float16* __restrict__ kh,
    _Float16* __restrict__ vh)
{
  const int lane = threadIdx.x & 31;
  const int wave = threadIdx.x >> 5;
  const int ln = lane & 15;
  const int hi = lane >> 4;
  const int nb = blockIdx.x * 256 + wave * 32;   // col base of this wave's pair
  const int m0 = blockIdx.y * 32;                // row base

  const int nc0 = nb + ln;
  const int nc1 = nb + 16 + ln;

  v8f c00 = zero8(), c01 = zero8(), c10 = zero8(), c11 = zero8();
  const _Float16* w0 = wh + (size_t)nc0 * CC;
  const _Float16* w1 = wh + (size_t)nc1 * CC;
  const _Float16* a0r = xh + (size_t)(m0 + ln) * CC;
  const _Float16* a1r = a0r + 16 * CC;

  for (int k0 = 0; k0 < CC; k0 += 32) {
    const v16h b0 = cat16(ld8h(w0 + k0 + hi * 8), ld8h(w0 + k0 + hi * 8 + 16));
    const v16h b1 = cat16(ld8h(w1 + k0 + hi * 8), ld8h(w1 + k0 + hi * 8 + 16));
    const _Float16* p0 = a0r + k0 + hi * 8;
    const v16h a0 = cat16(ld8h(p0), ld8h(p0 + 16));
    const _Float16* p1 = a1r + k0 + hi * 8;
    const v16h a1 = cat16(ld8h(p1), ld8h(p1 + 16));
    c00 = wmma16(a0, b0, c00);
    c01 = wmma16(a0, b1, c01);
    c10 = wmma16(a1, b0, c10);
    c11 = wmma16(a1, b1, c11);
  }

  // epilogue: bias + scatter into q/k/v [B,H,N,D]
  const float bn0 = bias[nc0];
  const float bn1 = bias[nc1];
  const int wh0 = nc0 >> 10, cc0 = nc0 & 1023, hh0 = cc0 >> 6, dd0 = cc0 & 63;
  const int wh1 = nc1 >> 10, cc1 = nc1 & 1023, hh1 = cc1 >> 6, dd1 = cc1 & 63;
  _Float16* base0 = (wh0 == 0) ? qh : (wh0 == 1) ? kh : vh;
  _Float16* base1 = (wh1 == 0) ? qh : (wh1 == 1) ? kh : vh;
#pragma unroll
  for (int r = 0; r < 8; r++) {
    int m = m0 + r + hi * 8;
    int bbi = m >> 11, tok = m & 2047;
    base0[((size_t)(bbi * HH + hh0) * NN + tok) * DD + dd0] = (_Float16)(c00[r] + bn0);
    base1[((size_t)(bbi * HH + hh1) * NN + tok) * DD + dd1] = (_Float16)(c01[r] + bn1);
    m += 16;
    bbi = m >> 11; tok = m & 2047;
    base0[((size_t)(bbi * HH + hh0) * NN + tok) * DD + dd0] = (_Float16)(c10[r] + bn0);
    base1[((size_t)(bbi * HH + hh1) * NN + tok) * DD + dd1] = (_Float16)(c11[r] + bn1);
  }
}

// ---------------------------------------------------------------------------
// Kernel 2: RoPE in place on q,k (f16 storage, f32 math). One thread handles
// the (d, d+32) pair for both q and k.
// ---------------------------------------------------------------------------
__global__ __launch_bounds__(256) void rope_kernel(
    const float* __restrict__ cosp, const float* __restrict__ sinp,
    _Float16* __restrict__ qh, _Float16* __restrict__ kh)
{
  const int t  = blockIdx.x * 256 + threadIdx.x;  // [0, 2^21)
  const int dd = t & 31;
  const int tok = (t >> 5) & 2047;
  const int bh  = t >> 16;                        // [0, 32)
  const size_t off = ((size_t)bh * NN + tok) * DD + dd;
  const int ci = tok * DD + dd;
  const float c1 = cosp[ci],      s1 = sinp[ci];
  const float c2 = cosp[ci + 32], s2 = sinp[ci + 32];

  const float q1 = (float)qh[off], q2 = (float)qh[off + 32];
  qh[off]      = (_Float16)(q1 * c1 - q2 * s1);
  qh[off + 32] = (_Float16)(q2 * c2 + q1 * s2);

  const float k1 = (float)kh[off], k2 = (float)kh[off + 32];
  kh[off]      = (_Float16)(k1 * c1 - k2 * s1);
  kh[off + 32] = (_Float16)(k2 * c2 + k1 * s2);
}

// ---------------------------------------------------------------------------
// Kernel 3: flash attention per (b,h). WG = 8 waves, each owns 16 query rows.
// K/V blocks of 32 rows staged in LDS (V transposed); online softmax with
// 16-lane shuffle reductions; P goes through per-wave LDS for the A-layout.
// ---------------------------------------------------------------------------
__global__ __launch_bounds__(256) void flash_attn_kernel(
    const _Float16* __restrict__ qh, const _Float16* __restrict__ kh,
    const _Float16* __restrict__ vh, _Float16* __restrict__ ah)
{
  __shared__ __align__(16) _Float16 Kt[32 * 64];      // K rows, row-major
  __shared__ __align__(16) _Float16 VT[64 * 32];      // V transposed: [d][kk]
  __shared__ __align__(16) _Float16 Pl[8][16 * 32];   // per-wave P tile

  const int lane = threadIdx.x & 31;
  const int wave = threadIdx.x >> 5;
  const int ln = lane & 15;
  const int hi = lane >> 4;
  const int bh = blockIdx.y;
  const int bb = bh >> 4;
  const int h  = bh & 15;
  const size_t bhOff = (size_t)bh * NN * DD;
  const int qRow0 = blockIdx.x * 128 + wave * 16;

  // Query A-fragments (d 0..31 and 32..63), loaded once.
  const _Float16* qp = qh + bhOff + (size_t)(qRow0 + ln) * DD + hi * 8;
  const v16h qa0 = cat16(ld8h(qp),      ld8h(qp + 16));
  const v16h qa1 = cat16(ld8h(qp + 32), ld8h(qp + 48));

  float m_run[8], l_run[8];
  v8f acc[4];
#pragma unroll
  for (int r = 0; r < 8; r++) { m_run[r] = -3.0e30f; l_run[r] = 0.f; }
#pragma unroll
  for (int t = 0; t < 4; t++) acc[t] = zero8();

  const int sr = threadIdx.x >> 3;  // staging row 0..31
  const int ss = threadIdx.x & 7;   // staging segment 0..7

  for (int j0 = 0; j0 < NN; j0 += 32) {
    // --- cooperative staging of 32 K rows and 32 V rows (V transposed) ---
    {
      const size_t srcOff = bhOff + (size_t)(j0 + sr) * DD + ss * 8;
      *(uint4*)(&Kt[sr * 64 + ss * 8]) = *(const uint4*)(kh + srcOff);
      uint4 vv = *(const uint4*)(vh + srcOff);
      const _Float16* ve = (const _Float16*)&vv;
#pragma unroll
      for (int e = 0; e < 8; e++) VT[(ss * 8 + e) * 32 + sr] = ve[e];
    }
    __syncthreads();

    // --- S = q @ k^T (two 16-wide key tiles) ---
    v8f S0 = zero8(), S1 = zero8();
    {
      const _Float16* kb0 = &Kt[ln * 64 + hi * 8];
      S0 = wmma16(qa0, cat16(ld8h(kb0),      ld8h(kb0 + 16)), S0);
      S0 = wmma16(qa1, cat16(ld8h(kb0 + 32), ld8h(kb0 + 48)), S0);
      const _Float16* kb1 = &Kt[(16 + ln) * 64 + hi * 8];
      S1 = wmma16(qa0, cat16(ld8h(kb1),      ld8h(kb1 + 16)), S1);
      S1 = wmma16(qa1, cat16(ld8h(kb1 + 32), ld8h(kb1 + 48)), S1);
    }

    // --- online softmax update (row stats live in 16-lane halves) ---
#pragma unroll
    for (int r = 0; r < 8; r++) {
      const float s0 = S0[r] * SCALE;
      const float s1 = S1[r] * SCALE;
      float mx = fmaxf(s0, s1);
#pragma unroll
      for (int off = 1; off < 16; off <<= 1)
        mx = fmaxf(mx, __shfl_xor(mx, off, 16));
      const float mn = fmaxf(m_run[r], mx);
      const float al = __expf(m_run[r] - mn);
      m_run[r] = mn;
      const float p0 = __expf(s0 - mn);
      const float p1 = __expf(s1 - mn);
      float rs = p0 + p1;
#pragma unroll
      for (int off = 1; off < 16; off <<= 1)
        rs += __shfl_xor(rs, off, 16);
      l_run[r] = l_run[r] * al + rs;
#pragma unroll
      for (int t = 0; t < 4; t++) acc[t][r] *= al;
      Pl[wave][(r + hi * 8) * 32 + ln]      = (_Float16)p0;
      Pl[wave][(r + hi * 8) * 32 + 16 + ln] = (_Float16)p1;
    }
    __syncthreads();

    // --- O += P @ V ---
    {
      const _Float16* pp = &Pl[wave][ln * 32 + hi * 8];
      const v16h pa = cat16(ld8h(pp), ld8h(pp + 16));
#pragma unroll
      for (int t = 0; t < 4; t++) {
        const _Float16* vb = &VT[(t * 16 + ln) * 32 + hi * 8];
        acc[t] = wmma16(pa, cat16(ld8h(vb), ld8h(vb + 16)), acc[t]);
      }
    }
    __syncthreads();
  }

  // --- finalize: divide by row sum, write [B,N,C] f16 ---
#pragma unroll
  for (int t = 0; t < 4; t++) {
#pragma unroll
    for (int r = 0; r < 8; r++) {
      const float ov = acc[t][r] / l_run[r];
      const int tok = qRow0 + r + hi * 8;
      const int col = h * DD + t * 16 + ln;
      ah[((size_t)(bb * NN + tok)) * CC + col] = (_Float16)ov;
    }
  }
}

// ---------------------------------------------------------------------------
// Kernel 4: output projection. ah[BN,C](f16) @ wph[C,C]^T(f16) + b -> f32 out.
// Same 2x2-tile-per-wave structure as kernel 1.
// ---------------------------------------------------------------------------
__global__ __launch_bounds__(256) void proj_gemm_kernel(
    const _Float16* __restrict__ a, const _Float16* __restrict__ wph,
    const float* __restrict__ bias, float* __restrict__ out)
{
  const int lane = threadIdx.x & 31;
  const int wave = threadIdx.x >> 5;
  const int ln = lane & 15;
  const int hi = lane >> 4;
  const int nb = blockIdx.x * 256 + wave * 32;
  const int m0 = blockIdx.y * 32;
  const int nc0 = nb + ln;
  const int nc1 = nb + 16 + ln;

  v8f c00 = zero8(), c01 = zero8(), c10 = zero8(), c11 = zero8();
  const _Float16* w0 = wph + (size_t)nc0 * CC;
  const _Float16* w1 = wph + (size_t)nc1 * CC;
  const _Float16* a0r = a + (size_t)(m0 + ln) * CC;
  const _Float16* a1r = a0r + 16 * CC;

  for (int k0 = 0; k0 < CC; k0 += 32) {
    const v16h b0 = cat16(ld8h(w0 + k0 + hi * 8), ld8h(w0 + k0 + hi * 8 + 16));
    const v16h b1 = cat16(ld8h(w1 + k0 + hi * 8), ld8h(w1 + k0 + hi * 8 + 16));
    const _Float16* p0 = a0r + k0 + hi * 8;
    const v16h a0 = cat16(ld8h(p0), ld8h(p0 + 16));
    const _Float16* p1 = a1r + k0 + hi * 8;
    const v16h a1 = cat16(ld8h(p1), ld8h(p1 + 16));
    c00 = wmma16(a0, b0, c00);
    c01 = wmma16(a0, b1, c01);
    c10 = wmma16(a1, b0, c10);
    c11 = wmma16(a1, b1, c11);
  }

  const float bn0 = bias[nc0];
  const float bn1 = bias[nc1];
#pragma unroll
  for (int r = 0; r < 8; r++) {
    const int m = m0 + r + hi * 8;
    out[(size_t)m * CC + nc0]        = c00[r] + bn0;
    out[(size_t)m * CC + nc1]        = c01[r] + bn1;
    out[(size_t)(m + 16) * CC + nc0] = c10[r] + bn0;
    out[(size_t)(m + 16) * CC + nc1] = c11[r] + bn1;
  }
}

// ---------------------------------------------------------------------------
extern "C" void kernel_launch(void* const* d_in, const int* in_sizes, int n_in,
                              void* d_out, int out_size, void* d_ws, size_t ws_size,
                              hipStream_t stream) {
  const float* x      = (const float*)d_in[0];
  const float* cosp   = (const float*)d_in[1];
  const float* sinp   = (const float*)d_in[2];
  const float* w_qkv  = (const float*)d_in[3];
  const float* b_qkv  = (const float*)d_in[4];
  const float* w_proj = (const float*)d_in[5];
  const float* b_proj = (const float*)d_in[6];
  float* out = (float*)d_out;

  const size_t perT   = (size_t)BB * HH * NN * DD;  // 4,194,304 halves (8 MB)
  const size_t nX     = (size_t)BB * NN * CC;       // 4,194,304
  const size_t nWqkv  = (size_t)3 * CC * CC;        // 3,145,728
  const size_t nWproj = (size_t)CC * CC;            // 1,048,576

  _Float16* qh  = (_Float16*)d_ws;
  _Float16* kh  = qh + perT;
  _Float16* vh  = kh + perT;
  _Float16* ah  = vh + perT;
  _Float16* xh  = ah + perT;
  _Float16* wqh = xh + nX;
  _Float16* wph = wqh + nWqkv;   // total ws use: 48 MB

  // 0) one-time f32 -> f16 conversion of activations and weights
  f32_to_f16_kernel<<<dim3((int)(nX / 8 / 256)), 256, 0, stream>>>(x, xh, (int)(nX / 8));
  f32_to_f16_kernel<<<dim3((int)(nWqkv / 8 / 256)), 256, 0, stream>>>(w_qkv, wqh, (int)(nWqkv / 8));
  f32_to_f16_kernel<<<dim3((int)(nWproj / 8 / 256)), 256, 0, stream>>>(w_proj, wph, (int)(nWproj / 8));

  // 1) QKV projection + bias -> f16 q/k/v in [B,H,N,D]
  qkv_gemm_kernel<<<dim3(3 * CC / 256, BB * NN / 32), 256, 0, stream>>>(
      xh, wqh, b_qkv, qh, kh, vh);

  // 2) RoPE on q,k in place
  rope_kernel<<<dim3((BB * HH * NN * (DD / 2)) / 256), 256, 0, stream>>>(
      cosp, sinp, qh, kh);

  // 3) Flash attention -> ah [B*N, C] f16
  flash_attn_kernel<<<dim3(NN / 128, BB * HH), 256, 0, stream>>>(qh, kh, vh, ah);

  // 4) Output projection + bias -> f32 d_out
  proj_gemm_kernel<<<dim3(CC / 256, BB * NN / 32), 256, 0, stream>>>(
      ah, wph, b_proj, out);
}
